// LSTMModel_8641474200315
// MI455X (gfx1250) — compile-verified, matrix operand order checked by reading, not run on hardware
//
#include <hip/hip_runtime.h>

// ---------------------------------------------------------------------------
// 3-layer LSTM (I=5, H=64, T=256, B=1024) + FC(64->1) for MI455X (gfx1250).
//
// Latency-bound sequential recurrence: each workgroup (512 thr = 16 wave32)
// owns 16 batch rows and runs all 3 layers x 256 timesteps locally.
// Each wave owns ONE 16-col gate tile; GEMMs are v_wmma_f32_16x16x32_f16
// (M=16 batch rows, N=16 gate cols, K=32 x2 for H=64). Weights are converted
// to f16 once and held in REGISTERS in B-operand lane layout (~80 VGPRs) for
// the whole time loop (<256 VGPRs total -> no vgpr-MSB churn). Gate columns
// are permuted to 4*unit+gate so each lane quad holds (i,f,g,o) of one hidden
// unit: gates are exchanged with 3 DPP quad_perm ops (no LDS), activations
// are branchless via tanh (native v_tanh_f32 when available), cell state c
// never leaves registers, and only h (f16) round-trips through 12KB of
// double-buffered LDS as the next A operand.
// ---------------------------------------------------------------------------

#define Bsz 1024
#define Tt  256
#define Hh  64
#define Iin 5

typedef _Float16 v16h __attribute__((ext_vector_type(16)));
typedef _Float16 v8h  __attribute__((ext_vector_type(8)));
typedef float    v8f  __attribute__((ext_vector_type(8)));

// --- single-instruction tanh when the gfx1250 builtin exists, else exp ---
__device__ __forceinline__ float tanh1(float x) {
#if __has_builtin(__builtin_amdgcn_tanhf)
  return __builtin_amdgcn_tanhf(x);
#elif __has_builtin(__builtin_amdgcn_tanh_f32)
  return __builtin_amdgcn_tanh_f32(x);
#else
  float e = __expf(2.0f * x);
  return 1.0f - 2.0f * __builtin_amdgcn_rcpf(e + 1.0f);
#endif
}

// --- quad_perm DPP cross-lane (VALU, no LDS). CTRL: 0xB1 = xor1, 0x4E = xor2.
template <int CTRL>
__device__ __forceinline__ float qperm(float v) {
#if __has_builtin(__builtin_amdgcn_mov_dpp)
  return __builtin_bit_cast(
      float, __builtin_amdgcn_mov_dpp(__builtin_bit_cast(int, v), CTRL, 0xF,
                                      0xF, true));
#else
  return __shfl_xor(v, (CTRL == 0xB1) ? 1 : 2, 32);
#endif
}

__device__ __forceinline__ v8f wmma_f16(v16h a, v16h b, v8f c) {
  // (neg_a, A, neg_b, B, c_mod, C, reuse_a, reuse_b)
  return __builtin_amdgcn_wmma_f32_16x16x32_f16(false, a, false, b, (short)0, c,
                                                false, false);
}

// Build a 32x16-f16 B operand (K x N tile, col = permuted gate col `c`) in the
// documented lane layout from row-major fp32 global weights [4H][kin].
// halves 0..7 = K(kb*32+hi+0..7), halves 8..15 = K(kb*32+16+hi+0..7)
__device__ __forceinline__ v16h load_bop(const float* __restrict__ w, int kin,
                                         int c, int kb, int hi) {
  int orow = (c & 3) * Hh + (c >> 2);  // permuted col -> original gate row
  const float* p = w + orow * kin + kb * 32 + hi;
  v16h r;
#pragma unroll
  for (int i = 0; i < 8; ++i) {
    r[i]     = (_Float16)p[i];
    r[i + 8] = (_Float16)p[i + 16];
  }
  return r;
}

// A operand (16x32 f16) from LDS row (64 halves) of hidden state; two 16-byte
// ds_load_b128 chunks per lane, matching the 16-bit A lane layout.
__device__ __forceinline__ v16h load_aop(const _Float16* row, int kb, int hi) {
  const _Float16* p = row + kb * 32 + hi;
  v8h lo = *(const v8h*)(p);
  v8h hc = *(const v8h*)(p + 16);
  return __builtin_shufflevector(lo, hc, 0, 1, 2, 3, 4, 5, 6, 7, 8, 9, 10, 11,
                                 12, 13, 14, 15);
}

// LSTM pointwise update for one 16x16 accumulator tile. The lane quad holds
// (i,f,g,o) of one hidden unit; gates exchanged via DPP quad_perm; c stays in
// registers (replicated across the quad); the g==0 lane stores h (f16) to LDS.
__device__ __forceinline__ void lstm_update(v8f acc, float* cst,
                                            _Float16* hcol_base, bool isG2,
                                            bool isOdd, bool isLo2, bool isG0,
                                            int hiRow) {
#pragma unroll
  for (int r = 0; r < 8; ++r) {
    float v = acc[r];
    // unified branchless activation: a = isG2 ? tanh(v) : sigmoid(v)
    //   sigmoid(v) = 0.5*tanh(0.5*v) + 0.5
    float xs = isG2 ? v : 0.5f * v;
    float tt = tanh1(xs);
    float a  = isG2 ? tt : __builtin_fmaf(tt, 0.5f, 0.5f);
    // quad exchange: lanes hold a = (i,f,g,o)
    float b1 = qperm<0xB1>(a);   // (f,i,o,g)
    float u  = isOdd ? b1 : a;   // (i,i,g,g)
    float w  = isOdd ? a  : b1;  // (f,f,o,o)
    float u2 = qperm<0x4E>(u);   // (g,g,i,i)
    float w2 = qperm<0x4E>(w);   // (o,o,f,f)
    float ig = u * u2;           // i*g in all lanes
    float fv = isLo2 ? w  : w2;  // f in all lanes
    float ov = isLo2 ? w2 : w;   // o in all lanes
    float cn = __builtin_fmaf(fv, cst[r], ig);
    cst[r] = cn;
    float hv = ov * tanh1(cn);
    if (isG0) hcol_base[(r + hiRow) * Hh] = (_Float16)hv;
  }
}

__global__ __launch_bounds__(512, 1) void lstm3_wmma_kernel(
    const float* __restrict__ x, const float* __restrict__ wih0,
    const float* __restrict__ whh0, const float* __restrict__ bih0,
    const float* __restrict__ bhh0, const float* __restrict__ wih1,
    const float* __restrict__ whh1, const float* __restrict__ bih1,
    const float* __restrict__ bhh1, const float* __restrict__ wih2,
    const float* __restrict__ whh2, const float* __restrict__ bih2,
    const float* __restrict__ bhh2, const float* __restrict__ wfc,
    const float* __restrict__ bfc, float* __restrict__ out) {
  // double-buffered hidden state per layer, f16, row-major [m][k]
  __shared__ _Float16 hb[3][2][16][Hh];

  const int  tid   = threadIdx.x;
  const int  wv    = tid >> 5;             // 16 waves
  const int  lane  = tid & 31;
  const int  m16   = lane & 15;
  const int  hi    = (lane < 16) ? 0 : 8;  // K-half select for A/B operands
  const int  hiRow = (lane < 16) ? 0 : 8;  // row offset for C/D tiles
  const int  g     = lane & 3;             // gate within quad (i,f,g,o)
  const bool isG2  = (g == 2);
  const bool isOdd = (g & 1) != 0;
  const bool isLo2 = (g < 2);
  const bool isG0  = (g == 0);
  const int  rbase = blockIdx.x * 16;

  // this wave owns permuted gate columns [wv*16, wv*16+16)
  const int c0 = wv * 16 + m16;
  const int j0 = c0 >> 2;  // hidden unit written back by g==0 lanes

  // ---- one-time: weights -> registers in B-operand layout ----
  v16h Bx;  // layer-0 input weights, single K=32 block, k<5 valid
  {
    v16h r = {};
    if (lane < 16) {
      int orow = (c0 & 3) * Hh + (c0 >> 2);
      const float* p = wih0 + orow * Iin;
#pragma unroll
      for (int i = 0; i < Iin; ++i) r[i] = (_Float16)p[i];
    }
    Bx = r;
  }
  v16h Bm[5][2];  // [matrix][kb]: 0=hh0 1=ih1 2=hh1 3=ih2 4=hh2
  {
    const float* wsrc[5] = {whh0, wih1, whh1, wih2, whh2};
#pragma unroll
    for (int m = 0; m < 5; ++m)
#pragma unroll
      for (int kb = 0; kb < 2; ++kb) Bm[m][kb] = load_bop(wsrc[m], Hh, c0, kb, hi);
  }

  // per-lane combined bias (b_ih + b_hh) per layer
  float bias[3];
  {
    const float* bi[3] = {bih0, bih1, bih2};
    const float* bh[3] = {bhh0, bhh1, bhh2};
    int orow = (c0 & 3) * Hh + (c0 >> 2);
#pragma unroll
    for (int l = 0; l < 3; ++l) bias[l] = bi[l][orow] + bh[l][orow];
  }

  // h(-1) = 0
  for (int idx = tid; idx < 3 * 2 * 16 * Hh; idx += 512)
    ((_Float16*)hb)[idx] = (_Float16)0.0f;

  // c(-1) = 0, in registers (replicated across each lane quad)
  float cst[3][8];
#pragma unroll
  for (int l = 0; l < 3; ++l)
#pragma unroll
    for (int r = 0; r < 8; ++r) cst[l][r] = 0.0f;

  __syncthreads();

  const float* xrow = x + (size_t)(rbase + m16) * Tt * Iin;  // lanes<16 only

  for (int t = 0; t < Tt; ++t) {
    const int cur = t & 1, prv = cur ^ 1;

    // ---------------- layer 0 ----------------
    v8f a0;
#pragma unroll
    for (int r = 0; r < 8; ++r) a0[r] = bias[0];
    v16h ax = {};  // x_t operand (16 rows x 5 valid K, zero padded)
    if (lane < 16) {
      const float* px = xrow + t * Iin;
#pragma unroll
      for (int i = 0; i < Iin; ++i) ax[i] = (_Float16)px[i];
    }
    a0 = wmma_f16(ax, Bx, a0);
#pragma unroll
    for (int kb = 0; kb < 2; ++kb)
      a0 = wmma_f16(load_aop(&hb[0][prv][m16][0], kb, hi), Bm[0][kb], a0);
    lstm_update(a0, cst[0], &hb[0][cur][0][j0], isG2, isOdd, isLo2, isG0, hiRow);
    __syncthreads();

    // ---------------- layer 1 ----------------
#pragma unroll
    for (int r = 0; r < 8; ++r) a0[r] = bias[1];
#pragma unroll
    for (int kb = 0; kb < 2; ++kb)
      a0 = wmma_f16(load_aop(&hb[0][cur][m16][0], kb, hi), Bm[1][kb], a0);
#pragma unroll
    for (int kb = 0; kb < 2; ++kb)
      a0 = wmma_f16(load_aop(&hb[1][prv][m16][0], kb, hi), Bm[2][kb], a0);
    lstm_update(a0, cst[1], &hb[1][cur][0][j0], isG2, isOdd, isLo2, isG0, hiRow);
    __syncthreads();

    // ---------------- layer 2 ----------------
#pragma unroll
    for (int r = 0; r < 8; ++r) a0[r] = bias[2];
#pragma unroll
    for (int kb = 0; kb < 2; ++kb)
      a0 = wmma_f16(load_aop(&hb[1][cur][m16][0], kb, hi), Bm[3][kb], a0);
#pragma unroll
    for (int kb = 0; kb < 2; ++kb)
      a0 = wmma_f16(load_aop(&hb[2][prv][m16][0], kb, hi), Bm[4][kb], a0);
    lstm_update(a0, cst[2], &hb[2][cur][0][j0], isG2, isOdd, isLo2, isG0, hiRow);
    __syncthreads();
  }

  // ---------------- final FC: out = h2(T-1) @ w_fc^T + b_fc ----------------
  if (tid < 16) {
    const _Float16* hrow = &hb[2][(Tt - 1) & 1][tid][0];
    float s = bfc[0];
#pragma unroll
    for (int k = 0; k < Hh; ++k) s += (float)hrow[k] * wfc[k];
    out[rbase + tid] = s;
  }
}

extern "C" void kernel_launch(void* const* d_in, const int* in_sizes, int n_in,
                              void* d_out, int out_size, void* d_ws,
                              size_t ws_size, hipStream_t stream) {
  const float* x    = (const float*)d_in[0];
  const float* wih0 = (const float*)d_in[1];
  const float* whh0 = (const float*)d_in[2];
  const float* bih0 = (const float*)d_in[3];
  const float* bhh0 = (const float*)d_in[4];
  const float* wih1 = (const float*)d_in[5];
  const float* whh1 = (const float*)d_in[6];
  const float* bih1 = (const float*)d_in[7];
  const float* bhh1 = (const float*)d_in[8];
  const float* wih2 = (const float*)d_in[9];
  const float* whh2 = (const float*)d_in[10];
  const float* bih2 = (const float*)d_in[11];
  const float* bhh2 = (const float*)d_in[12];
  const float* wfc  = (const float*)d_in[13];
  const float* bfc  = (const float*)d_in[14];
  float* out = (float*)d_out;

  lstm3_wmma_kernel<<<Bsz / 16, 512, 0, stream>>>(
      x, wih0, whh0, bih0, bhh0, wih1, whh1, bih1, bhh1, wih2, whh2, bih2,
      bhh2, wfc, bfc, out);
}
